// MultiHeadedAttention_72980084294316
// MI455X (gfx1250) — matrix operand order repaired and checked by reading, not compiled, not run on hardware
//
#include <hip/hip_runtime.h>
#include <hip/hip_bf16.h>

// ---------------------------------------------------------------------------
// MI455X (gfx1250, wave32) fused rel-pos causal attention, bf16 WMMA pipeline
// + TDM (tensor_load_to_lds) double-buffered K-tile staging.
// ---------------------------------------------------------------------------

typedef __attribute__((ext_vector_type(16))) __bf16         v16bf;
typedef __attribute__((ext_vector_type(8)))  float          v8f;
typedef __attribute__((ext_vector_type(16))) unsigned short v16u;
typedef __attribute__((ext_vector_type(8)))  unsigned short v8u;
typedef __attribute__((ext_vector_type(4)))  unsigned int   u32x4;
typedef __attribute__((ext_vector_type(8)))  int            i32x8;
typedef __attribute__((ext_vector_type(4)))  int            i32x4;

#if defined(__has_builtin)
#  if __has_builtin(__builtin_amdgcn_tensor_load_to_lds) && \
      __has_builtin(__builtin_amdgcn_s_wait_tensorcnt)
#    define HAVE_TDM 1
#  endif
#endif
#ifndef HAVE_TDM
#  define HAVE_TDM 0
#endif

#define WMMA_BF16(a, b, c) \
  __builtin_amdgcn_wmma_f32_16x16x32_bf16(false, (a), false, (b), (short)0, (c), false, false)

// round-to-nearest-even f32 -> bf16 bits
static __device__ __forceinline__ unsigned short f2bf(float f) {
  unsigned int u = __builtin_bit_cast(unsigned int, f);
  u += 0x7fffu + ((u >> 16) & 1u);
  return (unsigned short)(u >> 16);
}

// B fragment (32x16 bf16): lane col = lane&15, k = (lane>=16?16:0) + e -> 16 contiguous halves
static __device__ __forceinline__ v16bf loadB16(const unsigned short* p) {
  return __builtin_bit_cast(v16bf, *(const v16u*)p);
}
// A fragment (16x32 bf16): lane row = lane&15; k = h*16+e (p1 = p0 + 8)
static __device__ __forceinline__ v16bf loadA16(const unsigned short* p0,
                                                const unsigned short* p1) {
  v8u lo = *(const v8u*)p0;
  v8u hi = *(const v8u*)p1;
  v16u u;
#pragma unroll
  for (int e = 0; e < 8; ++e) { u[e] = lo[e]; u[8 + e] = hi[e]; }
  return __builtin_bit_cast(v16bf, u);
}

// ---------------------------------------------------------------------------
__global__ void mha_convert_bf16(const float* __restrict__ src,
                                 unsigned short* __restrict__ dst, int n) {
  int i = blockIdx.x * blockDim.x + threadIdx.x;
  if (i < n) dst[i] = f2bf(src[i]);
}

// erp:   [(16+S) x 64]  rows  0..15 zero, row 16+r = rel_pos[r]       (bf16)
// ervTp: [64 x (16+S)]  cols 0..15 zero, col 16+r = val_rel_pos[r][d] (bf16)
__global__ void mha_build_er(const float* __restrict__ rel,
                             const float* __restrict__ vrel,
                             unsigned short* __restrict__ erp,
                             unsigned short* __restrict__ ervTp) {
  int i = blockIdx.x * blockDim.x + threadIdx.x;
  if (i >= 1040 * 64) return;
  {
    int row = i / 64, d = i % 64, r = row - 16;
    erp[row * 64 + d] = (r >= 0) ? f2bf(rel[r * 64 + d]) : (unsigned short)0;
  }
  {
    int d = i / 1040, c = i % 1040, r = c - 16;
    ervTp[d * 1040 + c] = (r >= 0) ? f2bf(vrel[r * 64 + d]) : (unsigned short)0;
  }
}

// ---------------------------------------------------------------------------
// Projection GEMM: y = x @ W^T + b ; 1 wave computes a 16x16 tile, K=1024.
// which==0/1 -> store [BH, S, 64] (q/k); which==2 -> store v transposed [BH, 64, S].
__global__ __launch_bounds__(128) void mha_proj(
    const unsigned short* __restrict__ xq, const unsigned short* __restrict__ xk,
    const unsigned short* __restrict__ xv, const unsigned short* __restrict__ wq,
    const unsigned short* __restrict__ wk, const unsigned short* __restrict__ wv,
    const float* __restrict__ bq, const float* __restrict__ bk,
    const float* __restrict__ bv, unsigned short* __restrict__ qb,
    unsigned short* __restrict__ kbuf, unsigned short* __restrict__ vT) {
  const int lane = threadIdx.x & 31, wave = threadIdx.x >> 5;
  const int which = blockIdx.z;
  const unsigned short* X = (which == 0) ? xq : (which == 1) ? xk : xv;
  const unsigned short* W = (which == 0) ? wq : (which == 1) ? wk : wv;
  const float* bias       = (which == 0) ? bq : (which == 1) ? bk : bv;
  unsigned short* out     = (which == 0) ? qb : (which == 1) ? kbuf : vT;

  const int m0 = (blockIdx.x * 4 + wave) * 16;   // token-row tile (B*S = 2048 rows)
  const int n0 = blockIdx.y * 16;                // output-feature tile (D = 1024)
  const int col = lane & 15, hi = lane >> 4;

  const unsigned short* xrow = X + (m0 + col) * 1024;
  const unsigned short* wrow = W + (n0 + col) * 1024;  // B[k][n] = W[n][k]

  v8f acc = {};
#pragma unroll 4
  for (int c = 0; c < 32; ++c) {
    v16bf a = loadA16(xrow + c * 32 + hi * 16, xrow + c * 32 + 8 + hi * 16);
    v16bf b = loadB16(wrow + c * 32 + hi * 16);
    acc = WMMA_BF16(a, b, acc);
  }

  const int nG = n0 + col;
  const float bval = bias[nG];
  const int h = nG >> 6, dh = nG & 63;
#pragma unroll
  for (int v = 0; v < 8; ++v) {
    int mG = m0 + v + 8 * hi;
    int bb = mG >> 10, s = mG & 1023;
    unsigned short o = f2bf(acc[v] + bval);
    if (which < 2) out[(((bb << 4) + h) * 1024 + s) * 64 + dh] = o;
    else           out[(((bb << 4) + h) * 64 + dh) * 1024 + s] = o;
  }
}

// ---------------------------------------------------------------------------
// Flash attention with relative-position score bias and value term.
// One wave owns one (b, h, 16-row i-tile); streams 16-col j-tiles (causal).
// K tiles are DMA-staged into LDS by the Tensor Data Mover, double-buffered.
__global__ __launch_bounds__(128) void mha_flash(
    const unsigned short* __restrict__ qb, const unsigned short* __restrict__ kbuf,
    const unsigned short* __restrict__ vT, const unsigned short* __restrict__ erp,
    const unsigned short* __restrict__ ervTp, float* __restrict__ out) {
  __shared__ __align__(32) float          qeL[4][2][256];   // qe windows per wave
  __shared__ __align__(32) unsigned short pL[4][512];       // 16x32 P (bf16)
  __shared__ __align__(32) unsigned short skL[4][512];      // 16x32 skewed Prel
  __shared__ __align__(32) unsigned short kst[4][2][1024];  // K tile double buffer

  const int lane = threadIdx.x & 31, wave = threadIdx.x >> 5;
  const int wid = blockIdx.x * 4 + wave;
  const int it = wid & 63, bh = wid >> 6;     // 64 i-tiles per (b,h)
  const int i0 = it << 4;
  const int col = lane & 15, hi = lane >> 4;

  // Q tile as two K=32 A fragments (d = 0..31, 32..63)
  const unsigned short* qrow = qb + (bh * 1024 + i0 + col) * 64;
  const v16bf qa0 = loadA16(qrow + hi * 16,      qrow + 8 + hi * 16);
  const v16bf qa1 = loadA16(qrow + 32 + hi * 16, qrow + 40 + hi * 16);

  float* qe0 = qeL[wave][0];
  float* qe1 = qeL[wave][1];
  const int njt = it + 1;

  // --- TDM: DMA one 16x64 bf16 K tile (row stride 64) into LDS staging -----
  auto tdm_load_k = [&](int jt, int buf) {
#if HAVE_TDM
    unsigned long long gaddr =
        (unsigned long long)(const void*)(kbuf + ((size_t)bh * 1024 + (size_t)(jt << 4)) * 64);
    unsigned int lds = (unsigned int)(unsigned long long)(const void*)&kst[wave][buf][0];
    u32x4 g0;
    g0[0] = 1u;                                               // count=1, user mode
    g0[1] = lds;                                              // lds_addr
    g0[2] = (unsigned int)gaddr;                              // global_addr[31:0]
    g0[3] = ((unsigned int)(gaddr >> 32) & 0x01ffffffu) |     // global_addr[56:32]
            (2u << 30);                                       // type = 2 ("image")
    i32x8 g1;
    g1[0] = (int)(1u << 16);     // workgroup_mask=0, data_size=1 (2 bytes)
    g1[1] = (int)(64u << 16);    // tensor_dim0 = 64  (bits[79:48], low half)
    g1[2] = (int)(1024u << 16);  // dim0 hi=0 | tensor_dim1 = 1024 (bits[111:80], low half)
    g1[3] = (int)(64u << 16);    // dim1 hi=0 | tile_dim0 = 64
    g1[4] = 16;                  // tile_dim1 = 16 rows, tile_dim2 = 0
    g1[5] = 64;                  // tensor_dim0_stride = 64 elements
    g1[6] = 0;                   // stride0 hi / stride1 lo
    g1[7] = 0;                   // stride1 hi
    i32x4 gz4 = {0, 0, 0, 0};    // groups 2/3 unused (2-D tensor)
    i32x8 gz8 = {0, 0, 0, 0, 0, 0, 0, 0};
    __builtin_amdgcn_tensor_load_to_lds(g0, g1, gz4, gz4, gz8, 0);
#else
    // fallback: plain copy (32 ushorts per lane)
    const unsigned short* src = kbuf + ((size_t)bh * 1024 + (size_t)(jt << 4)) * 64 + lane * 32;
    v8u* dst = (v8u*)&kst[wave][buf][lane * 32];
    dst[0] = *(const v8u*)(src);
    dst[1] = *(const v8u*)(src + 8);
    dst[2] = *(const v8u*)(src + 16);
    dst[3] = *(const v8u*)(src + 24);
#endif
  };
  auto wait_k = [&]() {
#if HAVE_TDM
    __builtin_amdgcn_s_wait_tensorcnt(0);
#endif
  };

  // score tile: s[mi][nj] = (q.k + q.er[i-j]) / sqrt(64), causal-masked
  auto compute_val = [&](int jt, const unsigned short* kt, float* val8) {
    const int j0 = jt << 4;
    const int d0 = i0 - j0;
    // qe window: two 16-wide r subtiles covering r in [d0-16, d0+16)
#pragma unroll
    for (int sub = 0; sub < 2; ++sub) {
      int r0 = d0 - 16 + sub * 16;
      const unsigned short* er = erp + (16 + r0 + col) * 64 + hi * 16;
      v8f qe = {};
      qe = WMMA_BF16(qa0, loadB16(er), qe);
      qe = WMMA_BF16(qa1, loadB16(er + 32), qe);
      float* dst = sub ? qe1 : qe0;
#pragma unroll
      for (int v = 0; v < 8; ++v) dst[(v + 8 * hi) * 16 + col] = qe[v];
    }
    // raw scores q.k  (K tile staged in LDS, row jj = lane col)
    const unsigned short* krow = kt + col * 64;
    v8f s = {};
    s = WMMA_BF16(qa0, loadB16(krow + hi * 16), s);
    s = WMMA_BF16(qa1, loadB16(krow + 32 + hi * 16), s);
#pragma unroll
    for (int v = 0; v < 8; ++v) {
      int mi = v + 8 * hi;
      int r = d0 + mi - col;                       // relative distance i-j
      float bias = (r < d0) ? qe0[mi * 16 + (r - (d0 - 16))]
                            : qe1[mi * 16 + (r - d0)];
      float x = (s[v] + bias) * 0.125f;
      x = (d0 == 0 && col > mi) ? -3.0e38f : x;    // causal mask (diag tile only)
      val8[v] = x;
    }
  };

  // ---- pass 1: streaming max / sum -------------------------------------
  float m[8], l[8];
#pragma unroll
  for (int v = 0; v < 8; ++v) { m[v] = -3.0e38f; l[v] = 0.f; }
  tdm_load_k(0, 0);
  for (int jt = 0; jt < njt; ++jt) {
    wait_k();                                     // buffer jt ready
    if (jt + 1 < njt) tdm_load_k(jt + 1, (jt + 1) & 1);
    float val[8];
    compute_val(jt, kst[wave][jt & 1], val);
#pragma unroll
    for (int v = 0; v < 8; ++v) {
      float t = val[v];
      t = fmaxf(t, __shfl_xor(t, 1));
      t = fmaxf(t, __shfl_xor(t, 2));
      t = fmaxf(t, __shfl_xor(t, 4));
      t = fmaxf(t, __shfl_xor(t, 8));
      float mn = fmaxf(m[v], t);
      float p = __expf(val[v] - mn);
      p += __shfl_xor(p, 1);
      p += __shfl_xor(p, 2);
      p += __shfl_xor(p, 4);
      p += __shfl_xor(p, 8);
      l[v] = l[v] * __expf(m[v] - mn) + p;
      m[v] = mn;
    }
  }

  // ---- pass 2: P @ V  +  skew(P) @ ervT ---------------------------------
  v8f z[4] = {};
  unsigned short* myP = pL[wave];
  unsigned short* mySk = skL[wave];
  tdm_load_k(0, 0);
  for (int jt = 0; jt < njt; ++jt) {
    wait_k();
    if (jt + 1 < njt) tdm_load_k(jt + 1, (jt + 1) & 1);
    float val[8];
    compute_val(jt, kst[wave][jt & 1], val);
    if ((jt & 1) == 0) {             // fresh j-pair: clear 16x32 P tile
      v8u zz = {};
      *(v8u*)&myP[lane * 16] = zz;
      *(v8u*)&myP[lane * 16 + 8] = zz;
    }
    {                                 // clear skew tile every j-tile
      v8u zz = {};
      *(v8u*)&mySk[lane * 16] = zz;
      *(v8u*)&mySk[lane * 16 + 8] = zz;
    }
    unsigned short pb[8];
#pragma unroll
    for (int v = 0; v < 8; ++v) pb[v] = f2bf(__expf(val[v] - m[v]));
#pragma unroll
    for (int v = 0; v < 8; ++v) {
      int mi = v + 8 * hi;
      myP[mi * 32 + (jt & 1) * 16 + col] = pb[v];       // P[mi][jj]
      mySk[mi * 32 + 16 + mi - col] = pb[v];            // Prel[mi][16+mi-nj]
    }
    const int d0 = i0 - (jt << 4);
    // z += Prel @ ervT over r-window [d0-16, d0+16)  (padded, zeros where unused)
    const unsigned short* srow = &mySk[col * 32];
    v16bf pa = loadA16(srow + hi * 16, srow + 8 + hi * 16);
#pragma unroll
    for (int dt = 0; dt < 4; ++dt) {
      const unsigned short* eb = ervTp + (dt * 16 + col) * 1040 + d0 + hi * 16;
      z[dt] = WMMA_BF16(pa, loadB16(eb), z[dt]);
    }
    // z += P @ V once per 32-wide j-pair
    if ((jt & 1) == 1 || jt == njt - 1) {
      const unsigned short* prow = &myP[col * 32];
      v16bf pa2 = loadA16(prow + hi * 16, prow + 8 + hi * 16);
      int j0p = (jt & ~1) << 4;
#pragma unroll
      for (int dt = 0; dt < 4; ++dt) {
        const unsigned short* vb =
            vT + (bh * 64 + dt * 16 + col) * 1024 + j0p + hi * 16;
        z[dt] = WMMA_BF16(pa2, loadB16(vb), z[dt]);
      }
    }
  }

  // ---- epilogue: normalize, write [B, S, H*64] f32 ----------------------
  const int b = bh >> 4, h = bh & 15;
#pragma unroll
  for (int dt = 0; dt < 4; ++dt) {
#pragma unroll
    for (int v = 0; v < 8; ++v) {
      int i = i0 + v + 8 * hi;
      int d = dt * 16 + col;
      out[((b * 1024 + i) * 16 + h) * 64 + d] = z[dt][v] / l[v];
    }
  }
}

// ---------------------------------------------------------------------------
extern "C" void kernel_launch(void* const* d_in, const int* in_sizes, int n_in,
                              void* d_out, int out_size, void* d_ws, size_t ws_size,
                              hipStream_t stream) {
  (void)in_sizes; (void)n_in; (void)out_size; (void)ws_size;
  const float* query = (const float*)d_in[0];
  const float* key   = (const float*)d_in[1];
  const float* value = (const float*)d_in[2];
  const float* Wq    = (const float*)d_in[3];
  const float* bq    = (const float*)d_in[4];
  const float* Wk    = (const float*)d_in[5];
  const float* bk    = (const float*)d_in[6];
  const float* Wv    = (const float*)d_in[7];
  const float* bv    = (const float*)d_in[8];
  const float* rel   = (const float*)d_in[9];
  const float* vrel  = (const float*)d_in[10];

  const size_t NX = 2u * 1024u * 1024u;  // B*S*D
  const size_t NW = 1024u * 1024u;       // D*D
  unsigned short* ws   = (unsigned short*)d_ws;
  unsigned short* xq   = ws;        ws += NX;
  unsigned short* xk   = ws;        ws += NX;
  unsigned short* xv   = ws;        ws += NX;
  unsigned short* wqb  = ws;        ws += NW;
  unsigned short* wkb  = ws;        ws += NW;
  unsigned short* wvb  = ws;        ws += NW;
  unsigned short* qb   = ws;        ws += NX;   // [BH, S, 64] bf16
  unsigned short* kbuf = ws;        ws += NX;   // [BH, S, 64] bf16
  unsigned short* vT   = ws;        ws += NX;   // [BH, 64, S] bf16
  unsigned short* erp  = ws;        ws += 1040u * 64u;   // padded er
  unsigned short* ervTp= ws;        ws += 64u * 1040u;   // padded ervT

  const int thr = 256;
  mha_convert_bf16<<<(int)((NX + thr - 1) / thr), thr, 0, stream>>>(query, xq, (int)NX);
  mha_convert_bf16<<<(int)((NX + thr - 1) / thr), thr, 0, stream>>>(key,   xk, (int)NX);
  mha_convert_bf16<<<(int)((NX + thr - 1) / thr), thr, 0, stream>>>(value, xv, (int)NX);
  mha_convert_bf16<<<(int)((NW + thr - 1) / thr), thr, 0, stream>>>(Wq, wqb, (int)NW);
  mha_convert_bf16<<<(int)((NW + thr - 1) / thr), thr, 0, stream>>>(Wk, wkb, (int)NW);
  mha_convert_bf16<<<(int)((NW + thr - 1) / thr), thr, 0, stream>>>(Wv, wvb, (int)NW);
  mha_build_er<<<(1040 * 64 + thr - 1) / thr, thr, 0, stream>>>(rel, vrel, erp, ervTp);

  // 2048 token-rows / 16 = 128 M-tiles (4 waves/block), 64 N-tiles, 3 matrices
  mha_proj<<<dim3(32, 64, 3), 128, 0, stream>>>(xq, xk, xv, wqb, wkb, wvb,
                                                bq, bk, bv, qb, kbuf, vT);
  // 2 * 16 * 64 i-tiles = 2048 waves / 4 per block
  mha_flash<<<512, 128, 0, stream>>>(qb, kbuf, vT, erp, ervTp, (float*)d_out);
}
